// _UserModel_82806969467402
// MI455X (gfx1250) — matrix operand order repaired and checked by reading, not compiled, not run on hardware
//
#include <hip/hip_runtime.h>
#include <hip/hip_bf16.h>

// ---------- WMMA vector types (gfx1250, wave32) ----------
typedef __attribute__((ext_vector_type(16))) __bf16   v16bf;
typedef __attribute__((ext_vector_type(8)))  float    v8f;
typedef __attribute__((ext_vector_type(8)))  unsigned int v8u;

#define L_SEQ   200
#define TPAD    208      // 13 tiles * 16
#define D64     64

// workspace fragment offsets (dwords); each 16x16x32 B-fragment = 32 lanes * 8 dwords = 256 dwords
#define OFF_G1  0        // gW1: 4 ktiles x 4 ntiles
#define OFF_G2  4096     // gW2: 2 x 2
#define OFF_G3  5120     // gW3: 1 x 4
#define OFF_A1  6144     // aW1: 4 x 4
#define OFF_A2  10240    // aW2: 2 x 2
#define FRAG_DWORDS 11264

// ---------- helpers ----------
__device__ __forceinline__ unsigned short f2bf(float f) {
    unsigned int u = __builtin_bit_cast(unsigned int, f);
    unsigned int r = u + 0x7FFFu + ((u >> 16) & 1u);   // round-to-nearest-even
    return (unsigned short)(r >> 16);
}
__device__ __forceinline__ unsigned int pack2(float lo, float hi) {
    return (unsigned int)f2bf(lo) | ((unsigned int)f2bf(hi) << 16);
}
__device__ __forceinline__ float bf2f(unsigned int bits16) {
    return __builtin_bit_cast(float, bits16 << 16);
}
// wave-local LDS RAW fence (same-wave DS ops are in-order; this pins the compiler too)
__device__ __forceinline__ void wave_lds_fence() {
    __builtin_amdgcn_wave_barrier();
    asm volatile("s_wait_dscnt 0" ::: "memory");
    __builtin_amdgcn_wave_barrier();
}

// ---------- weight packing: row-major (K x N) f32 -> WMMA bf16 B-operand fragments ----------
// B layout (16-bit, 32x16): lane = {half, n16}; element e -> Kloc = e + 16*half, col = n16
__global__ void pack_b_frags(const float* __restrict__ W, int N, unsigned int* __restrict__ out) {
    const int kt = blockIdx.x, nt = blockIdx.y, ntiles = gridDim.y;
    const int lane = threadIdx.x;
    const int n16 = lane & 15, hs = lane >> 4;
    unsigned int* dst = out + (size_t)(kt * ntiles + nt) * 256 + lane * 8;
    const int n = nt * 16 + n16;
#pragma unroll
    for (int j = 0; j < 8; ++j) {
        int k0 = kt * 32 + (2 * j)     + 16 * hs;
        int k1 = kt * 32 + (2 * j + 1) + 16 * hs;
        dst[j] = pack2(W[(size_t)k0 * N + n], W[(size_t)k1 * N + n]);
    }
}

// ---------- one WMMA MLP layer: A (bf16 pairs, LDS) x W-frags (LDS) -> f32 D staged to LDS ----------
// A layout (16-bit, 16x32): lane row m = lane&15; dword j -> K-pair 8*half+j
__device__ __forceinline__ void run_layer(const unsigned int* __restrict__ Arow,
                                          const unsigned int* __restrict__ wbase,
                                          const float* __restrict__ bias,
                                          float* __restrict__ F,
                                          int KT, int NT, int width, bool do_relu, int lane) {
    const int n16 = lane & 15, hs = lane >> 4;
    for (int nt = 0; nt < NT; ++nt) {
        const float bv = bias[nt * 16 + n16];
        v8f c;
#pragma unroll
        for (int i = 0; i < 8; ++i) c[i] = bv;
        for (int kt = 0; kt < KT; ++kt) {
            v8u ad;
#pragma unroll
            for (int j = 0; j < 8; ++j) {
                int P = 8 * hs + j;
                ad[j] = Arow[kt * 16 + P];
            }
            v8u bd;
            const unsigned int* bp = wbase + (kt * NT + nt) * 256 + lane * 8;
#pragma unroll
            for (int j = 0; j < 8; ++j) bd[j] = bp[j];
            v16bf a = __builtin_bit_cast(v16bf, ad);
            v16bf b = __builtin_bit_cast(v16bf, bd);
            c = __builtin_amdgcn_wmma_f32_16x16x32_bf16(false, a, false, b, (short)0, c, false, false);
        }
        // D layout: lane col n = lane&15, vgpr r -> row m = r + 8*half
#pragma unroll
        for (int r = 0; r < 8; ++r) {
            float v = c[r];
            if (do_relu) v = fmaxf(v, 0.f);
            F[(r + 8 * hs) * width + nt * 16 + n16] = v;
        }
    }
}

// ---------- fused per-batch-row kernel ----------
__global__ __launch_bounds__(256) void saer_fused_kernel(
    const int* __restrict__ uids, const int* __restrict__ u_item_pad,
    const float* __restrict__ user_tab, const float* __restrict__ item_tab,
    const float* __restrict__ rate_tab,
    const float* __restrict__ gb1, const float* __restrict__ gb2, const float* __restrict__ gb3,
    const float* __restrict__ ab1, const float* __restrict__ ab2,
    const float* __restrict__ aW3, const float* __restrict__ ab3,
    const float* __restrict__ hW1, const float* __restrict__ hb1,
    const float* __restrict__ hW2, const float* __restrict__ hb2,
    const unsigned int* __restrict__ frags, float* __restrict__ out) {

    __shared__ unsigned int wlds[FRAG_DWORDS];  // all WMMA B-fragments, resident in LDS (45 KB)
    __shared__ unsigned int bufA[8][16][64];    // per-wave A-operand (bf16 pairs), up to K=128
    __shared__ float        bufF[8][16][64];    // per-wave f32 D staging, up to N=64
    __shared__ float        maskw[8][16];
    __shared__ unsigned int x_lds[TPAD][32];    // x_ia, packed bf16 pairs (64 feats)
    __shared__ float        alpha_lds[TPAD];
    __shared__ float        pu[64];
    __shared__ float        agg_part[8][64];
    __shared__ float        agg[64];
    __shared__ float        h1s[64];
    __shared__ float        red_sum;

    const int b    = blockIdx.x;
    const int tid  = threadIdx.x;
    const int lane = tid & 31;
    const int wave = tid >> 5;

    // ---- phase A: one-shot fragment preload to LDS + user embedding + alpha init ----
    {
        const uint4* src = (const uint4*)frags;
        uint4*       dst = (uint4*)wlds;
        for (int i = tid; i < FRAG_DWORDS / 4; i += 256) dst[i] = src[i];
    }
    if (tid < 64) pu[tid] = user_tab[(size_t)uids[b] * D64 + tid];
    for (int i = tid; i < TPAD; i += 256) alpha_lds[i] = 0.f;
    if (tid == 0) red_sum = 0.f;
    __syncthreads();

    // ---- phase B: per-wave 16-token tiles, WMMA MLPs (all operands LDS-resident) ----
    unsigned int (*A)[64] = bufA[wave];
    float        (*F)[64] = bufF[wave];
    const unsigned int* Arow = &A[lane & 15][0];

    for (int iter = 0; iter < 2; ++iter) {
        const int tile = wave + iter * 8;          // 0..15 (13..15 are dummies, fully masked)
        const int t0   = tile * 16;

        // gather: concat(item_emb, rate_emb) -> bf16 A pairs (K = 0..127)
        for (int m = 0; m < 16; ++m) {
            const int t = t0 + m;
            int iid = 0, rid = 0;
            if (t < L_SEQ) {
                const size_t base = ((size_t)b * L_SEQ + t) * 2;
                iid = u_item_pad[base];
                rid = u_item_pad[base + 1];
            }
            if (lane == 0) maskw[wave][m] = (t < L_SEQ && iid > 0) ? 1.f : 0.f;
            const float* qa = item_tab + (size_t)iid * D64;
            A[m][lane] = pack2(qa[2 * lane], qa[2 * lane + 1]);
            const float* er = rate_tab + (size_t)rid * D64;
            A[m][32 + lane] = pack2(er[2 * lane], er[2 * lane + 1]);
        }
        wave_lds_fence();

        // gMLP: 128 -> 64 (relu) -> 32 (relu) -> 64
        run_layer(Arow, wlds + OFF_G1, gb1, &F[0][0], 4, 4, 64, true, lane);
        wave_lds_fence();
        for (int idx = lane; idx < 512; idx += 32) {
            int m = idx >> 5, p = idx & 31;
            A[m][p] = pack2(F[m][2 * p], F[m][2 * p + 1]);
        }
        wave_lds_fence();
        run_layer(Arow, wlds + OFF_G2, gb2, &F[0][0], 2, 2, 32, true, lane);
        wave_lds_fence();
        for (int idx = lane; idx < 256; idx += 32) {
            int m = idx >> 4, p = idx & 15;
            A[m][p] = pack2(F[m][2 * p], F[m][2 * p + 1]);
        }
        wave_lds_fence();
        run_layer(Arow, wlds + OFF_G3, gb3, &F[0][0], 1, 4, 64, false, lane);  // x_ia (no relu)
        wave_lds_fence();

        // x_ia -> A pairs 0..31 (aMLP input lo) + persist to x_lds
        for (int idx = lane; idx < 512; idx += 32) {
            int m = idx >> 5, p = idx & 31;
            unsigned int v = pack2(F[m][2 * p], F[m][2 * p + 1]);
            A[m][p] = v;
            int t = t0 + m;
            if (t < TPAD) x_lds[t][p] = v;
        }
        // p_i = mask * user_emb -> A pairs 32..63
        for (int idx = lane; idx < 512; idx += 32) {
            int m = idx >> 5, p = idx & 31;
            float msk = maskw[wave][m];
            A[m][32 + p] = pack2(msk * pu[2 * p], msk * pu[2 * p + 1]);
        }
        wave_lds_fence();

        // aMLP: 128 -> 64 (relu) -> 32 (relu) -> 1
        run_layer(Arow, wlds + OFF_A1, ab1, &F[0][0], 4, 4, 64, true, lane);
        wave_lds_fence();
        for (int idx = lane; idx < 512; idx += 32) {
            int m = idx >> 5, p = idx & 31;
            A[m][p] = pack2(F[m][2 * p], F[m][2 * p + 1]);
        }
        wave_lds_fence();
        run_layer(Arow, wlds + OFF_A2, ab2, &F[0][0], 2, 2, 32, true, lane);
        wave_lds_fence();

        // final 32->1 dot per token: lane pair (m, m+16) splits K
        float s = 0.f;
        {
            const int m = lane & 15, kb = (lane >> 4) * 16;
#pragma unroll
            for (int k = 0; k < 16; ++k) s += F[m][kb + k] * aW3[kb + k];
        }
        s += __shfl_xor(s, 16);
        if (lane < 16) {
            const int t = t0 + lane;
            if (t < TPAD) alpha_lds[t] = maskw[wave][lane] * __expf(s + ab3[0]);
        }
        wave_lds_fence();
    }
    __syncthreads();

    // ---- phase C: masked softmax denominator ----
    float part = (tid < TPAD) ? alpha_lds[tid] : 0.f;
#pragma unroll
    for (int off = 16; off; off >>= 1) part += __shfl_down(part, off);
    if (lane == 0) atomicAdd(&red_sum, part);
    __syncthreads();
    const float inv = 1.0f / (red_sum + 1e-10f);

    // ---- phase D: agg = sum_t alpha[t] * x_ia[t] ----
    {
        const int p = tid & 31, chunk = tid >> 5;
        float a0 = 0.f, a1 = 0.f;
        for (int t = chunk; t < TPAD; t += 8) {
            float al = alpha_lds[t];
            unsigned int v = x_lds[t][p];
            a0 += al * bf2f(v & 0xFFFFu);
            a1 += al * bf2f(v >> 16);
        }
        agg_part[chunk][2 * p]     = a0;
        agg_part[chunk][2 * p + 1] = a1;
    }
    __syncthreads();
    if (tid < 64) {
        float s = 0.f;
#pragma unroll
        for (int c = 0; c < 8; ++c) s += agg_part[c][tid];
        agg[tid] = s * inv;
    }
    __syncthreads();

    // ---- phase E: head MLP 64 -> 64 (relu) -> 64 (relu) ----
    if (tid < 64) {
        float s = hb1[tid];
        for (int k = 0; k < 64; ++k) s += agg[k] * hW1[k * 64 + tid];
        h1s[tid] = fmaxf(s, 0.f);
    }
    __syncthreads();
    if (tid < 64) {
        float s = hb2[tid];
        for (int k = 0; k < 64; ++k) s += h1s[k] * hW2[k * 64 + tid];
        out[(size_t)b * 64 + tid] = fmaxf(s, 0.f);
    }
}

extern "C" void kernel_launch(void* const* d_in, const int* in_sizes, int n_in,
                              void* d_out, int out_size, void* d_ws, size_t ws_size,
                              hipStream_t stream) {
    const int*   uids       = (const int*)d_in[0];
    const int*   u_item_pad = (const int*)d_in[1];
    const float* user_tab   = (const float*)d_in[2];
    const float* item_tab   = (const float*)d_in[3];
    const float* rate_tab   = (const float*)d_in[4];
    const float* gW1 = (const float*)d_in[5];  const float* gb1 = (const float*)d_in[6];
    const float* gW2 = (const float*)d_in[7];  const float* gb2 = (const float*)d_in[8];
    const float* gW3 = (const float*)d_in[9];  const float* gb3 = (const float*)d_in[10];
    const float* aW1 = (const float*)d_in[11]; const float* ab1 = (const float*)d_in[12];
    const float* aW2 = (const float*)d_in[13]; const float* ab2 = (const float*)d_in[14];
    const float* aW3 = (const float*)d_in[15]; const float* ab3 = (const float*)d_in[16];
    const float* hW1 = (const float*)d_in[17]; const float* hb1 = (const float*)d_in[18];
    const float* hW2 = (const float*)d_in[19]; const float* hb2 = (const float*)d_in[20];

    unsigned int* frags = (unsigned int*)d_ws;

    // pack weights into WMMA B-operand bf16 fragments (deterministic, every call)
    pack_b_frags<<<dim3(4, 4), 32, 0, stream>>>(gW1, 64, frags + OFF_G1);
    pack_b_frags<<<dim3(2, 2), 32, 0, stream>>>(gW2, 32, frags + OFF_G2);
    pack_b_frags<<<dim3(1, 4), 32, 0, stream>>>(gW3, 64, frags + OFF_G3);
    pack_b_frags<<<dim3(4, 4), 32, 0, stream>>>(aW1, 64, frags + OFF_A1);
    pack_b_frags<<<dim3(2, 2), 32, 0, stream>>>(aW2, 32, frags + OFF_A2);

    saer_fused_kernel<<<dim3(4096), dim3(256), 0, stream>>>(
        uids, u_item_pad, user_tab, item_tab, rate_tab,
        gb1, gb2, gb3, ab1, ab2, aW3, ab3,
        hW1, hb1, hW2, hb2, frags, (float*)d_out);
}
